// SparseSelfAttention_27831388078818
// MI455X (gfx1250) — matrix operand order repaired and checked by reading, not compiled, not run on hardware
//
#include <hip/hip_runtime.h>

// ---------------------------------------------------------------------------
// SparseSelfAttention for MI455X (gfx1250, wave32, WMMA).
// Reference: Q=xWq+bq, K=xWk+bk, V=xWv+bv; softmax(QK^T)V (mask is a no-op).
// B=4, S=4096, D=256, fp32 in/out. Internally bf16 WMMA + f32 accumulate.
// ---------------------------------------------------------------------------

typedef __attribute__((ext_vector_type(16))) __bf16 v16bf;
typedef __attribute__((ext_vector_type(8)))  __bf16 v8bf;
typedef __attribute__((ext_vector_type(8)))  float  v8f;
typedef __attribute__((ext_vector_type(4)))  int    i4vec;
typedef i4vec __attribute__((address_space(1)))* as1_i4p;  // global int4*
typedef i4vec __attribute__((address_space(3)))* as3_i4p;  // LDS int4*

#define SEQ  4096
#define DIM  256
#define NBAT 4
#define MROWS (NBAT * SEQ) // 16384

// gfx1250 async global->LDS copy (ASYNCcnt path), guarded per-toolchain.
#if defined(__has_builtin)
# if __has_builtin(__builtin_amdgcn_global_load_async_to_lds_b128)
#  define USE_ASYNC_LDS 1
# endif
#endif

static __device__ __forceinline__ void wait_asynccnt0() {
#if defined(__has_builtin) && __has_builtin(__builtin_amdgcn_s_wait_asynccnt)
  __builtin_amdgcn_s_wait_asynccnt(0);
#else
  asm volatile("s_wait_asynccnt 0x0" ::: "memory");
#endif
}

static __device__ __forceinline__ v8f wmma_bf16(v16bf a, v16bf b, v8f c) {
  // D = A(16x32) * B(32x16) + C(16x16), f32 accumulate.
  return __builtin_amdgcn_wmma_f32_16x16x32_bf16(
      /*neg_a=*/false, a, /*neg_b=*/false, b,
      /*c_mod=*/(short)0, c, /*reuse_a=*/false, /*reuse_b=*/false);
}

// ---------------------------------------------------------------------------
// Kernel 1: fused QKV projection.  out[r, c] = sum_k X[r,k]*W[k,c] + bias[c]
// Grid: (MROWS/128, DIM/64, 3={q,k,v}); 256 threads = 8 waves.
// Each wave: 16 rows x 64 cols (4 WMMA accumulators), K-loop in chunks of 32.
// ---------------------------------------------------------------------------
__global__ __launch_bounds__(256) void qkv_gemm_kernel(
    const float* __restrict__ X,
    const float* __restrict__ Wq, const float* __restrict__ Bq,
    const float* __restrict__ Wk, const float* __restrict__ Bk,
    const float* __restrict__ Wv, const float* __restrict__ Bv,
    __bf16* __restrict__ Qo, __bf16* __restrict__ Ko, __bf16* __restrict__ Vo)
{
  const float* W; const float* bias; __bf16* dst;
  if (blockIdx.z == 0)      { W = Wq; bias = Bq; dst = Qo; }
  else if (blockIdx.z == 1) { W = Wk; bias = Bk; dst = Ko; }
  else                      { W = Wv; bias = Bv; dst = Vo; }

  // W chunk staged transposed: Wt[c][k] = W[k0+k][colBase+c]  (B-layout friendly)
  __shared__ __align__(64) __bf16 Wt[64][32];

  const int tid  = threadIdx.x;
  const int wave = tid >> 5;
  const int lane = tid & 31;
  const int half = lane >> 4;   // which 16-lane half
  const int l16  = lane & 15;
  const int rowBase = blockIdx.x * 128 + wave * 16;
  const int colBase = blockIdx.y * 64;

  const v8f zerov = {0.f,0.f,0.f,0.f,0.f,0.f,0.f,0.f};
  v8f acc[4];
  #pragma unroll
  for (int c = 0; c < 4; ++c) acc[c] = zerov;

  for (int k0 = 0; k0 < DIM; k0 += 32) {
    __syncthreads();
    // stage 32x64 W chunk (transposed, f32 -> bf16); coalesced over c
    #pragma unroll
    for (int i = 0; i < 8; ++i) {
      int idx = i * 256 + tid;           // 0..2047
      int k   = idx >> 6;                // 0..31
      int c   = idx & 63;                // 0..63
      Wt[c][k] = (__bf16)W[(k0 + k) * DIM + colBase + c];
    }
    __syncthreads();

    // A tile (16x32): lane = row, K contiguous half*16 + 0..15
    v16bf a;
    {
      const float* xp = X + (size_t)(rowBase + l16) * DIM + k0 + half * 16;
      #pragma unroll
      for (int e = 0; e < 8; ++e) { a[e] = (__bf16)xp[e]; a[8 + e] = (__bf16)xp[8 + e]; }
    }
    // B tiles: lane = column (colBase+16c+l16), elements = K half*16 + 0..15 (contiguous)
    #pragma unroll
    for (int c = 0; c < 4; ++c) {
      v16bf b = *(const v16bf*)(&Wt[c * 16 + l16][half * 16]);
      acc[c] = wmma_bf16(a, b, acc[c]);
    }
  }

  // bias + store bf16 (C-layout: VGPR j -> row j + 8*half, lane -> col l16)
  #pragma unroll
  for (int c = 0; c < 4; ++c) {
    const int col = colBase + c * 16 + l16;
    const float bb = bias[col];
    #pragma unroll
    for (int j = 0; j < 8; ++j) {
      const int r = rowBase + j + half * 8;
      dst[(size_t)r * DIM + col] = (__bf16)(acc[c][j] + bb);
    }
  }
}

// ---------------------------------------------------------------------------
// Kernel 2: flash attention.  Grid: (SEQ/64, NBAT); 128 threads = 4 waves.
// Each wave owns 16 query rows; loop over 32-key blocks with online softmax.
// ---------------------------------------------------------------------------
__global__ __launch_bounds__(128) void flash_attn_kernel(
    const __bf16* __restrict__ Q, const __bf16* __restrict__ K,
    const __bf16* __restrict__ V, float* __restrict__ O)
{
  __shared__ __align__(64) __bf16 Kl[32][DIM];   // K block, row-major   (16 KB)
  __shared__ __align__(64) __bf16 Vt[DIM][32];   // V block, transposed  (16 KB)
  __shared__ __align__(64) __bf16 Pl[4][16][32]; // per-wave P staging   ( 4 KB)

  const int tid  = threadIdx.x;
  const int wave = tid >> 5;
  const int lane = tid & 31;
  const int half = lane >> 4;
  const int l16  = lane & 15;
  const int bidx = blockIdx.y;
  const int qRow0 = blockIdx.x * 64 + wave * 16;

  const __bf16* Qb = Q + (size_t)bidx * SEQ * DIM;
  const __bf16* Kb = K + (size_t)bidx * SEQ * DIM;
  const __bf16* Vb = V + (size_t)bidx * SEQ * DIM;

  // Load this wave's Q rows into 8 A-layout tiles (16x32 each, covering D=256)
  v16bf qa[8];
  {
    const __bf16* qp = Qb + (size_t)(qRow0 + l16) * DIM;
    #pragma unroll
    for (int t = 0; t < 8; ++t) {
      v8bf lo = *(const v8bf*)(qp + t * 32 + half * 16);
      v8bf hi = *(const v8bf*)(qp + t * 32 + half * 16 + 8);
      #pragma unroll
      for (int e = 0; e < 8; ++e) { qa[t][e] = lo[e]; qa[t][8 + e] = hi[e]; }
    }
  }

  const v8f zerov = {0.f,0.f,0.f,0.f,0.f,0.f,0.f,0.f};
  float mrow[8], lrow[8];
  #pragma unroll
  for (int j = 0; j < 8; ++j) { mrow[j] = -3.0e38f; lrow[j] = 0.0f; }
  v8f oacc[16];
  #pragma unroll
  for (int c = 0; c < 16; ++c) oacc[c] = zerov;

  for (int j0 = 0; j0 < SEQ; j0 += 32) {
    __syncthreads();
    // stage K (row-major, async global->LDS when available) and V (transposed)
    #pragma unroll
    for (int i = 0; i < 8; ++i) {
      int idx = i * 128 + tid;          // 0..1023
      int kk  = idx >> 5;               // key row 0..31
      int cc  = (idx & 31) * 8;         // col start
#ifdef USE_ASYNC_LDS
      __builtin_amdgcn_global_load_async_to_lds_b128(
          (as1_i4p)(void*)(Kb + (size_t)(j0 + kk) * DIM + cc),
          (as3_i4p)(void*)&Kl[kk][cc],
          /*offset=*/0, /*cpol=*/0);
#else
      v8bf kv = *(const v8bf*)(Kb + (size_t)(j0 + kk) * DIM + cc);
      *(v8bf*)(&Kl[kk][cc]) = kv;
#endif
      v8bf vv = *(const v8bf*)(Vb + (size_t)(j0 + kk) * DIM + cc);
      #pragma unroll
      for (int e = 0; e < 8; ++e) Vt[cc + e][kk] = vv[e];
    }
    if (j0 + 32 < SEQ) {  // global_prefetch_b8 for next block
      __builtin_prefetch(Kb + (size_t)(j0 + 32) * DIM + lane * 64, 0, 1);
      __builtin_prefetch(Vb + (size_t)(j0 + 32) * DIM + lane * 64, 0, 1);
    }
#ifdef USE_ASYNC_LDS
    wait_asynccnt0();
#endif
    __syncthreads();

    // scores S = Q * K^T : two 16x16 tiles (h = key group), 8 WMMA each over D
    v8f sc[2]; sc[0] = zerov; sc[1] = zerov;
    #pragma unroll
    for (int h = 0; h < 2; ++h) {
      #pragma unroll
      for (int t = 0; t < 8; ++t) {
        // B operand: lane = key (h*16+l16), elements = d (t*32 + half*16 + 0..15)
        v16bf bm = *(const v16bf*)(&Kl[h * 16 + l16][t * 32 + half * 16]);
        sc[h] = wmma_bf16(qa[t], bm, sc[h]);
      }
    }

    // online softmax; row j+8*half is replicated across the 16 lanes of a half
    float alpha[8];
    #pragma unroll
    for (int j = 0; j < 8; ++j) {
      float mx = fmaxf(sc[0][j], sc[1][j]);
      #pragma unroll
      for (int m = 8; m >= 1; m >>= 1) mx = fmaxf(mx, __shfl_xor(mx, m, 32));
      const float mnew = fmaxf(mrow[j], mx);
      alpha[j] = __expf(mrow[j] - mnew);
      const float p0 = __expf(sc[0][j] - mnew);
      const float p1 = __expf(sc[1][j] - mnew);
      sc[0][j] = p0; sc[1][j] = p1;
      float rs = p0 + p1;
      #pragma unroll
      for (int m = 8; m >= 1; m >>= 1) rs += __shfl_xor(rs, m, 32);
      lrow[j] = lrow[j] * alpha[j] + rs;
      mrow[j] = mnew;
    }

    // P: C-layout -> A-layout via per-wave LDS round-trip (bf16)
    #pragma unroll
    for (int h = 0; h < 2; ++h) {
      #pragma unroll
      for (int j = 0; j < 8; ++j)
        Pl[wave][j + half * 8][h * 16 + l16] = (__bf16)sc[h][j];
    }
    asm volatile("s_wait_dscnt 0x0" ::: "memory"); // cross-lane LDS RAW in-wave

    // rescale running output by alpha (per-row)
    #pragma unroll
    for (int c = 0; c < 16; ++c) {
      #pragma unroll
      for (int j = 0; j < 8; ++j) oacc[c][j] *= alpha[j];
    }

    v16bf pa;
    {
      const __bf16* pp = &Pl[wave][l16][half * 16];
      v8bf lo = *(const v8bf*)pp;         // keys half*16 .. +7
      v8bf hi = *(const v8bf*)(pp + 8);   // keys half*16+8 .. +15
      #pragma unroll
      for (int e = 0; e < 8; ++e) { pa[e] = lo[e]; pa[8 + e] = hi[e]; }
    }

    // O += P * V : 16 column tiles, B operand from transposed V (contiguous keys)
    #pragma unroll
    for (int c = 0; c < 16; ++c) {
      v16bf bm = *(const v16bf*)(&Vt[c * 16 + l16][half * 16]);
      oacc[c] = wmma_bf16(pa, bm, oacc[c]);
    }
  }

  // epilogue: normalize by l and store fp32
  float invl[8];
  #pragma unroll
  for (int j = 0; j < 8; ++j) invl[j] = 1.0f / lrow[j];
  float* Ob = O + (size_t)bidx * SEQ * DIM;
  #pragma unroll
  for (int c = 0; c < 16; ++c) {
    const int col = c * 16 + l16;
    #pragma unroll
    for (int j = 0; j < 8; ++j) {
      const int r = qRow0 + j + half * 8;
      Ob[(size_t)r * DIM + col] = oacc[c][j] * invl[j];
    }
  }
}

// ---------------------------------------------------------------------------
extern "C" void kernel_launch(void* const* d_in, const int* in_sizes, int n_in,
                              void* d_out, int out_size, void* d_ws, size_t ws_size,
                              hipStream_t stream) {
  (void)in_sizes; (void)n_in; (void)out_size; (void)ws_size;
  const float* x  = (const float*)d_in[0];
  const float* Wq = (const float*)d_in[1];
  const float* bq = (const float*)d_in[2];
  const float* Wk = (const float*)d_in[3];
  const float* bk = (const float*)d_in[4];
  const float* Wv = (const float*)d_in[5];
  const float* bv = (const float*)d_in[6];

  // workspace: bf16 Q, K, V each [16384, 256] -> 24 MB total
  __bf16* Qw = (__bf16*)d_ws;
  __bf16* Kw = Qw + (size_t)MROWS * DIM;
  __bf16* Vw = Kw + (size_t)MROWS * DIM;

  dim3 gA(MROWS / 128, DIM / 64, 3);
  qkv_gemm_kernel<<<gA, 256, 0, stream>>>(x, Wq, bq, Wk, bk, Wv, bv, Qw, Kw, Vw);

  dim3 gB(SEQ / 64, NBAT);
  flash_attn_kernel<<<gB, 128, 0, stream>>>(Qw, Kw, Vw, (float*)d_out);
}
